// LSTM_9002251453229
// MI455X (gfx1250) — matrix operand order repaired and checked by reading, not compiled, not run on hardware
//
#include <hip/hip_runtime.h>
#include <hip/hip_fp16.h>
#include <stdint.h>

typedef __attribute__((ext_vector_type(16))) _Float16 v16h;
typedef __attribute__((ext_vector_type(8)))  _Float16 v8h;
typedef __attribute__((ext_vector_type(8)))  float    v8f;
typedef __attribute__((ext_vector_type(4)))  float    v4f;

static constexpr int B_ = 64, T_ = 512, E_ = 256, H_ = 256, G_ = 1024; // G = 4H
static constexpr int EPAD = 264; // padded LDS row stride (halves); 528B = 33*16, keeps 16B alignment
#define DROP_P_ 0.3f

// ---- async global->LDS path (CDNA5) with compile-safe fallback --------------
#if defined(__has_builtin)
# if __has_builtin(__builtin_amdgcn_global_load_async_to_lds_b128) && \
     __has_builtin(__builtin_amdgcn_s_wait_asynccnt)
#  define USE_ASYNC_LDS 1
# endif
#endif
#ifndef USE_ASYNC_LDS
# define USE_ASYNC_LDS 0
#endif

#if USE_ASYNC_LDS
typedef int i4 __attribute__((vector_size(16)));
typedef __attribute__((address_space(1))) i4 i4_glob;   // AS1 (printed as __device__)
typedef __attribute__((address_space(3))) i4 i4_lds;    // AS3 (printed as __shared__)
#endif

__device__ __forceinline__ void copy_xg_tile(const _Float16* __restrict__ gsrc,
                                             _Float16* __restrict__ ldst,
                                             int tid) {
  // 16 rows x 1024 halves = 32KB contiguous; 512 threads x 64B each
#if USE_ASYNC_LDS
  const size_t off = (size_t)tid * 64;
  i4_glob* g = (i4_glob*)(uintptr_t)((const char*)gsrc + off); // generic->AS1: same address
  i4_lds*  l = (i4_lds*)(uintptr_t)((char*)ldst + off);        // generic->AS3: low 32 bits = LDS offset
#pragma unroll
  for (int i = 0; i < 4; ++i)
    __builtin_amdgcn_global_load_async_to_lds_b128(g + i, l + i, 0, 0);
#else
  const uint4* s = (const uint4*)gsrc;
  uint4* d = (uint4*)ldst;
#pragma unroll
  for (int i = 0; i < 4; ++i) d[tid * 4 + i] = s[tid * 4 + i];
#endif
}

__device__ __forceinline__ void wait_xg() {
#if USE_ASYNC_LDS
  __builtin_amdgcn_s_wait_asynccnt(0);
#endif
}

__device__ __forceinline__ float sigf(float x) {
  return 1.0f / (1.0f + __expf(-x));
}

// ---- kernel 0: WeightDrop + f32->f16 conversion of both weight matrices -----
__global__ void prep_weights(const float* __restrict__ W_ih,
                             const float* __restrict__ W_hh,
                             const float* __restrict__ dmask,
                             _Float16* __restrict__ Wih_h,
                             _Float16* __restrict__ Whh_h) {
  int i = blockIdx.x * blockDim.x + threadIdx.x;
  if (i < G_ * H_) {
    Wih_h[i] = (_Float16)W_ih[i];
    float keep = (dmask[i] >= DROP_P_) ? (1.0f / (1.0f - DROP_P_)) : 0.0f;
    Whh_h[i] = (_Float16)(W_hh[i] * keep);
  }
}

// ---- kernel 1: xg[t*B+b, 4H] = emb[tok]·W_ihᵀ + (b_ih + b_hh), via WMMA -----
__global__ __launch_bounds__(256) void xg_gemm(const int* __restrict__ tokens,
                                               const float* __restrict__ emb,
                                               const _Float16* __restrict__ Wih_h,
                                               const float* __restrict__ b_ih,
                                               const float* __restrict__ b_hh,
                                               _Float16* __restrict__ xg) {
  __shared__ __align__(16) _Float16 Asm[16][EPAD];
  const int tid = threadIdx.x;
  const int mtile = blockIdx.x;       // 2048 tiles of 16 rows over M = T*B
  const int t = mtile >> 2;
  const int b0 = (mtile & 3) << 4;

  // gather 16 embedding rows, f32 -> f16 into LDS (A staging)
  {
    int r = tid >> 4;
    int c0 = (tid & 15) << 4;
    int tok = tokens[(b0 + r) * T_ + t];
    const float* src = emb + (size_t)tok * E_ + c0;
#pragma unroll
    for (int i = 0; i < 16; i += 4) {
      v4f v = *(const v4f*)(src + i);
      Asm[r][c0 + i + 0] = (_Float16)v.x;
      Asm[r][c0 + i + 1] = (_Float16)v.y;
      Asm[r][c0 + i + 2] = (_Float16)v.z;
      Asm[r][c0 + i + 3] = (_Float16)v.w;
    }
  }
  __syncthreads();

  const int wave = tid >> 5;
  const int lane = tid & 31;
  const int lrow = lane & 15;
  const int hi = (lane & 16) ? 1 : 0;

  for (int nt = wave * 8; nt < wave * 8 + 8; ++nt) {
    const int n = nt * 16 + lrow;
    const _Float16* brow = Wih_h + (size_t)n * E_;
    v8f acc = {};
#pragma unroll
    for (int kc = 0; kc < 8; ++kc) {
      const int ka = kc * 32 + hi * 16;  // A: K={0..15} or {16..31}
      v8h a0 = *(const v8h*)(&Asm[lrow][ka]);
      v8h a1 = *(const v8h*)(&Asm[lrow][ka + 8]);
      v16h A = __builtin_shufflevector(a0, a1, 0, 1, 2, 3, 4, 5, 6, 7,
                                       8, 9, 10, 11, 12, 13, 14, 15);
      const int kb = kc * 32 + hi * 16;  // B: contiguous 16-half run of W row n
      v16h Bv = *(const v16h*)(brow + kb);
      acc = __builtin_amdgcn_wmma_f32_16x16x32_f16(false, A, false, Bv,
                                                   (short)0, acc, false, false);
    }
    const float bias = b_ih[n] + b_hh[n];
    const int mb = b0 + hi * 8;
#pragma unroll
    for (int r = 0; r < 8; ++r)
      xg[((size_t)t * B_ + mb + r) * G_ + n] = (_Float16)(acc[r] + bias);
  }
}

// ---- kernel 2: sequential LSTM scan; 4 WGs x 16 batch rows, state in LDS ----
__global__ __launch_bounds__(512) void lstm_scan(const _Float16* __restrict__ xg,
                                                 const _Float16* __restrict__ Whh_h,
                                                 const float* __restrict__ lin_W,
                                                 const float* __restrict__ lin_b,
                                                 float* __restrict__ out) {
  __shared__ __align__(16) _Float16 xgbuf[2][16 * G_];   // 64KB, double-buffered
  __shared__ __align__(16) _Float16 hbuf[2][16][EPAD];   // h state (f16 A operand)
  __shared__ __align__(16) float cbuf[16][H_ + 4];       // c state (f32)

  const int tid = threadIdx.x;
  const int b0 = blockIdx.x * 16;
  const int wave = tid >> 5;
  const int lane = tid & 31;
  const int lrow = lane & 15;
  const int hi = (lane & 16) ? 1 : 0;
  const int jcol = wave * 16 + lrow;   // each wave owns one 16-col slice of H

  for (int i = tid; i < 16 * EPAD; i += 512) {
    hbuf[0][0][i] = (_Float16)0.0f;
    hbuf[1][0][i] = (_Float16)0.0f;
  }
  for (int i = tid; i < 16 * (H_ + 4); i += 512) ((float*)cbuf)[i] = 0.0f;

  copy_xg_tile(xg + (size_t)b0 * G_, &xgbuf[0][0], tid);
  wait_xg();
  __syncthreads();

  float* hfin = (float*)&xgbuf[0][0];  // reused for final h (safe: t=511 reads xgbuf[1])

  const _Float16* rI = Whh_h + (size_t)(jcol)*H_;
  const _Float16* rF = Whh_h + (size_t)(jcol + 256) * H_;
  const _Float16* rG = Whh_h + (size_t)(jcol + 512) * H_;
  const _Float16* rO = Whh_h + (size_t)(jcol + 768) * H_;

  for (int t = 0; t < T_; ++t) {
    const int buf = t & 1;
    if (t + 1 < T_)  // overlap next tile's DMA with this step's WMMA
      copy_xg_tile(xg + ((size_t)(t + 1) * B_ + b0) * G_, &xgbuf[buf ^ 1][0], tid);

    v8f ai = {}, af = {}, ag = {}, ao = {};
#pragma unroll
    for (int kc = 0; kc < 8; ++kc) {
      const int ka = kc * 32 + hi * 16;
      v8h a0 = *(const v8h*)(&hbuf[buf][lrow][ka]);
      v8h a1 = *(const v8h*)(&hbuf[buf][lrow][ka + 8]);
      v16h A = __builtin_shufflevector(a0, a1, 0, 1, 2, 3, 4, 5, 6, 7,
                                       8, 9, 10, 11, 12, 13, 14, 15);
      const int kb = kc * 32 + hi * 16;
      ai = __builtin_amdgcn_wmma_f32_16x16x32_f16(false, A, false, *(const v16h*)(rI + kb), (short)0, ai, false, false);
      af = __builtin_amdgcn_wmma_f32_16x16x32_f16(false, A, false, *(const v16h*)(rF + kb), (short)0, af, false, false);
      ag = __builtin_amdgcn_wmma_f32_16x16x32_f16(false, A, false, *(const v16h*)(rG + kb), (short)0, ag, false, false);
      ao = __builtin_amdgcn_wmma_f32_16x16x32_f16(false, A, false, *(const v16h*)(rO + kb), (short)0, ao, false, false);
    }

    const int mb = hi * 8;
#pragma unroll
    for (int r = 0; r < 8; ++r) {
      const int m = mb + r;
      float xi = (float)xgbuf[buf][m * G_ + jcol];
      float xf = (float)xgbuf[buf][m * G_ + jcol + 256];
      float xgg = (float)xgbuf[buf][m * G_ + jcol + 512];
      float xo = (float)xgbuf[buf][m * G_ + jcol + 768];
      float I = sigf(ai[r] + xi);
      float F = sigf(af[r] + xf);
      float Gg = tanhf(ag[r] + xgg);
      float O = sigf(ao[r] + xo);
      float c = F * cbuf[m][jcol] + I * Gg;   // (m,jcol) uniquely owned: no race
      cbuf[m][jcol] = c;
      float h = O * tanhf(c);
      hbuf[buf ^ 1][m][jcol] = (_Float16)h;   // consumed after the barrier
      if (t == T_ - 1) hfin[m * H_ + jcol] = h;
    }
    wait_xg();
    __syncthreads();
  }

  // final projection [16,256]x[256,2] + 2-way softmax (tiny)
  if (tid < 16) {
    const int m = tid;
    float l0 = lin_b[0], l1 = lin_b[1];
    for (int k = 0; k < H_; ++k) {
      float h = hfin[m * H_ + k];
      l0 += h * lin_W[k];
      l1 += h * lin_W[256 + k];
    }
    float mx = fmaxf(l0, l1);
    float e0 = __expf(l0 - mx), e1 = __expf(l1 - mx);
    float inv = 1.0f / (e0 + e1);
    out[(b0 + m) * 2 + 0] = e0 * inv;
    out[(b0 + m) * 2 + 1] = e1 * inv;
  }
}

extern "C" void kernel_launch(void* const* d_in, const int* in_sizes, int n_in,
                              void* d_out, int out_size, void* d_ws, size_t ws_size,
                              hipStream_t stream) {
  const int* tokens = (const int*)d_in[0];
  const float* emb = (const float*)d_in[1];
  const float* W_ih = (const float*)d_in[2];
  const float* W_hh = (const float*)d_in[3];
  const float* b_ih = (const float*)d_in[4];
  const float* b_hh = (const float*)d_in[5];
  const float* lin_W = (const float*)d_in[6];
  const float* lin_b = (const float*)d_in[7];
  const float* dmask = (const float*)d_in[8];
  float* out = (float*)d_out;

  char* ws = (char*)d_ws;
  _Float16* Wih_h = (_Float16*)ws;                              // 512 KB
  _Float16* Whh_h = (_Float16*)(ws + (size_t)512 * 1024);       // 512 KB
  _Float16* xg = (_Float16*)(ws + (size_t)1024 * 1024);         // 64 MB: [T][B][4H] f16

  prep_weights<<<(G_ * H_ + 255) / 256, 256, 0, stream>>>(W_ih, W_hh, dmask, Wih_h, Whh_h);
  xg_gemm<<<2048, 256, 0, stream>>>(tokens, emb, Wih_h, b_ih, b_hh, xg);
  lstm_scan<<<4, 512, 0, stream>>>(xg, Whh_h, lin_W, lin_b, out);
}
